// Encoder_eicu_34591666601997
// MI455X (gfx1250) — matrix operand order repaired and checked by reading, not compiled, not run on hardware
//
#include <hip/hip_runtime.h>
#include <hip/hip_bf16.h>

typedef __attribute__((ext_vector_type(16))) _Float16 v16h;
typedef __attribute__((ext_vector_type(8)))  _Float16 v8h;
typedef __attribute__((ext_vector_type(8)))  float    v8f;

#define B_    32
#define T_    40
#define C_    24
#define D_    256
#define N_    960      /* T_*C_ */
#define CT_   73       /* 3*C_ + 1 */
#define SCALE_ 0.0625f /* 1/sqrt(256+1e-7) rounds to 0.0625 in f32 */

// ---------------------------------------------------------------------------
// K1: time feature  tf = (1 - tanh^2(st*W_sel + b_sel)) @ W_time + b_time
// ---------------------------------------------------------------------------
__global__ __launch_bounds__(64)
void k1_timefeat(const float* __restrict__ st, const float* __restrict__ W_sel,
                 const float* __restrict__ b_sel, const float* __restrict__ W_time,
                 const float* __restrict__ b_time, float* __restrict__ tf32)
{
    const int bt = blockIdx.x, j = threadIdx.x;   // 64 threads
    __shared__ float h[64];
    float u  = st[bt] * W_sel[j] + b_sel[j];
    float th = tanhf(u);
    h[j] = 1.f - th * th;
    __syncthreads();
    float acc[4];
#pragma unroll
    for (int k = 0; k < 4; k++) acc[k] = b_time[j + 64 * k];
    for (int jj = 0; jj < 64; jj++) {
        float hj = h[jj];
#pragma unroll
        for (int k = 0; k < 4; k++) acc[k] += hj * W_time[jj * D_ + j + 64 * k];
    }
#pragma unroll
    for (int k = 0; k < 4; k++) tf32[(size_t)bt * D_ + j + 64 * k] = acc[k];
}

// ---------------------------------------------------------------------------
// K2: embedding gather * mask, gating attention + sigmoid gate, write gated
//     values twice: row-major (gated_h) and D-major transposed (gated_t)
// ---------------------------------------------------------------------------
__global__ __launch_bounds__(256)
void k2_gate(const int* __restrict__ med_codes, const int* __restrict__ lab_codes,
             const int* __restrict__ diag_codes,
             const float* __restrict__ med_mask, const float* __restrict__ lab_mask,
             const float* __restrict__ diag_mask,
             const float* __restrict__ emb_med, const float* __restrict__ emb_lab,
             const float* __restrict__ emb_diag,
             const float* __restrict__ Wg_med, const float* __restrict__ bg_med,
             const float* __restrict__ Wg_lab, const float* __restrict__ bg_lab,
             const float* __restrict__ Wg_diag, const float* __restrict__ bg_diag,
             const float* __restrict__ tf32,
             _Float16* __restrict__ gated_h, _Float16* __restrict__ gated_t)
{
    const int bt = blockIdx.x;
    const int b  = bt / T_, t = bt % T_;
    const int s  = blockIdx.y;
    const int d  = threadIdx.x;

    const int*   codes; const float* maskc; const float* emb;
    const float* Wg;    const float* bg;
    if (s == 0)      { codes = med_codes;  maskc = med_mask;  emb = emb_med;  Wg = Wg_med;  bg = bg_med;  }
    else if (s == 1) { codes = lab_codes;  maskc = lab_mask;  emb = emb_lab;  Wg = Wg_lab;  bg = bg_lab;  }
    else             { codes = diag_codes; maskc = diag_mask; emb = emb_diag; Wg = Wg_diag; bg = bg_diag; }

    __shared__ float outl[C_ * D_];
    __shared__ float tfl[D_];
    __shared__ float gl[D_];
    __shared__ float scl[C_];
    __shared__ int   codeL[C_];

    if (threadIdx.x < C_) codeL[threadIdx.x] = codes[(size_t)bt * C_ + threadIdx.x];
    tfl[d] = tf32[(size_t)bt * D_ + d];
    __syncthreads();

    for (int c = 0; c < C_; c++) {
        float mk = maskc[(size_t)bt * C_ + c];
        outl[c * D_ + d] = emb[(size_t)codeL[c] * D_ + d] * mk;
    }
    __syncthreads();

    const int wave = threadIdx.x >> 5, lane = threadIdx.x & 31;
    for (int c = wave * 3; c < wave * 3 + 3; c++) {
        float p = 0.f;
#pragma unroll
        for (int k = 0; k < 8; k++) { int dd = lane + 32 * k; p += tfl[dd] * outl[c * D_ + dd]; }
#pragma unroll
        for (int off = 16; off > 0; off >>= 1) p += __shfl_xor(p, off, 32);
        scl[c] = p * SCALE_;
    }
    __syncthreads();

    float mx = -1e30f;
    for (int c = 0; c < C_; c++) mx = fmaxf(mx, scl[c]);
    float den = 0.f;
    for (int c = 0; c < C_; c++) den += __expf(scl[c] - mx);
    float inv = 1.f / den;

    float g = 0.f;
    for (int c = 0; c < C_; c++) g += __expf(scl[c] - mx) * inv * outl[c * D_ + d];
    gl[d] = g;
    __syncthreads();

    float acc = bg[d];
    for (int j = 0; j < D_; j++) acc += gl[j] * Wg[j * D_ + d];
    float gg = 1.f / (1.f + __expf(-acc));

    _Float16 vals[C_];
    size_t rowbase = ((size_t)(s * B_ + b) * N_ + (size_t)t * C_) * D_;
    for (int c = 0; c < C_; c++) {
        float v = gg * outl[c * D_ + d];
        vals[c] = (_Float16)v;
        gated_h[rowbase + (size_t)c * D_ + d] = vals[c];
    }
    _Float16* tp = gated_t + ((size_t)(s * B_ + b) * D_ + d) * N_ + (size_t)t * C_;
#pragma unroll
    for (int c8 = 0; c8 < 3; c8++) {
        v8h pk;
#pragma unroll
        for (int q = 0; q < 8; q++) pk[q] = vals[c8 * 8 + q];
        *(v8h*)(tp + c8 * 8) = pk;
    }
}

// ---------------------------------------------------------------------------
// K3: flash-attention over N=960 slots, D=256, WMMA f16->f32.
//     One wave per 16-row Q tile; 32-key blocks; online softmax.
// ---------------------------------------------------------------------------
__global__ __launch_bounds__(32)
void k3_stream_attn(const _Float16* __restrict__ gated_h,
                    const _Float16* __restrict__ gated_t,
                    float* __restrict__ satt)
{
    const int mt = blockIdx.x;   // Q tile (0..59)
    const int sb = blockIdx.y;   // stream*32 + batch (0..95)
    const _Float16* __restrict__ KQ = gated_h + (size_t)sb * N_ * D_;
    const _Float16* __restrict__ Vt = gated_t + (size_t)sb * D_ * N_;
    float* __restrict__ Out = satt + (size_t)sb * N_ * D_;

    const int lane = threadIdx.x;
    const int lm = lane & 15, lg = lane >> 4;
    const int q0 = mt * 16;

    __shared__ __align__(64) _Float16 Plds[16 * 32];

    // Q fragments in WMMA A layout: lane -> row m=lm; half h -> k=lg*16+h
    v16h qf[8];
    {
        const _Float16* qrow = KQ + (size_t)(q0 + lm) * D_;
#pragma unroll
        for (int c = 0; c < 8; c++) {
            v8h lo = *(const v8h*)(qrow + 32 * c + lg * 16);
            v8h hi = *(const v8h*)(qrow + 32 * c + lg * 16 + 8);
            qf[c] = __builtin_shufflevector(lo, hi, 0,1,2,3,4,5,6,7,8,9,10,11,12,13,14,15);
        }
    }

    v8f o[16];
#pragma unroll
    for (int jj = 0; jj < 16; jj++) {
        v8f z = {0.f,0.f,0.f,0.f,0.f,0.f,0.f,0.f};
        o[jj] = z;
    }
    float mrow[8], lrow[8];
#pragma unroll
    for (int r = 0; r < 8; r++) { mrow[r] = -1e30f; lrow[r] = 0.f; }

    for (int kb = 0; kb < N_; kb += 32) {
        v8f s0 = {0.f,0.f,0.f,0.f,0.f,0.f,0.f,0.f};
        v8f s1 = {0.f,0.f,0.f,0.f,0.f,0.f,0.f,0.f};
#pragma unroll
        for (int c = 0; c < 8; c++) {
            const _Float16* kbase = KQ + (size_t)(kb + lm) * D_ + 32 * c + lg * 16;
            v16h b0 = *(const v16h*)kbase;
            v16h b1 = *(const v16h*)(kbase + 16 * D_);
            s0 = __builtin_amdgcn_wmma_f32_16x16x32_f16(false, qf[c], false, b0, (short)0, s0, false, false);
            s1 = __builtin_amdgcn_wmma_f32_16x16x32_f16(false, qf[c], false, b1, (short)0, s1, false, false);
        }
        float pa0[8], pa1[8];
#pragma unroll
        for (int r = 0; r < 8; r++) {
            float a  = s0[r] * SCALE_;
            float bb = s1[r] * SCALE_;
            float mx = fmaxf(a, bb);
#pragma unroll
            for (int off = 1; off < 16; off <<= 1) mx = fmaxf(mx, __shfl_xor(mx, off, 32));
            float mnew  = fmaxf(mrow[r], mx);
            float alpha = __expf(mrow[r] - mnew);
            float e0 = __expf(a - mnew);
            float e1 = __expf(bb - mnew);
            float rs = e0 + e1;
#pragma unroll
            for (int off = 1; off < 16; off <<= 1) rs += __shfl_xor(rs, off, 32);
            lrow[r] = lrow[r] * alpha + rs;
            mrow[r] = mnew;
            pa0[r] = e0; pa1[r] = e1;
#pragma unroll
            for (int jj = 0; jj < 16; jj++) o[jj][r] *= alpha;
        }
#pragma unroll
        for (int r = 0; r < 8; r++) {
            Plds[(r + 8 * lg) * 32 + lm]      = (_Float16)pa0[r];
            Plds[(r + 8 * lg) * 32 + 16 + lm] = (_Float16)pa1[r];
        }
        __syncthreads();
        v8h plo = *(const v8h*)(Plds + lm * 32 + lg * 16);
        v8h phi = *(const v8h*)(Plds + lm * 32 + lg * 16 + 8);
        v16h pa = __builtin_shufflevector(plo, phi, 0,1,2,3,4,5,6,7,8,9,10,11,12,13,14,15);
        __syncthreads();
#pragma unroll
        for (int jj = 0; jj < 16; jj++) {
            v16h vb = *(const v16h*)(Vt + (size_t)(16 * jj + lm) * N_ + kb + lg * 16);
            o[jj] = __builtin_amdgcn_wmma_f32_16x16x32_f16(false, pa, false, vb, (short)0, o[jj], false, false);
        }
    }

    float invl[8];
#pragma unroll
    for (int r = 0; r < 8; r++) invl[r] = 1.f / lrow[r];
#pragma unroll
    for (int jj = 0; jj < 16; jj++) {
#pragma unroll
        for (int r = 0; r < 8; r++) {
            Out[(size_t)(q0 + r + 8 * lg) * D_ + 16 * jj + lm] = o[jj][r] * invl[r];
        }
    }
}

// ---------------------------------------------------------------------------
// K4a: merged 73-slot attention via WMMA. 5 waves; rows padded 73->96.
//      Writes O2 row-major (mout_h, pitch 80 rows) and transposed
//      (mout_t, pitch 96 keys so PV K-chunks stay in-bounds).
// ---------------------------------------------------------------------------
#define K4A_LDS (96*256*2 + 256*96*2 + 5*16*96*2)

__global__ __launch_bounds__(160)
void k4a_merge_attn(const float* __restrict__ satt, const float* __restrict__ tf32,
                    _Float16* __restrict__ mout_h, _Float16* __restrict__ mout_t)
{
    extern __shared__ __align__(64) char smem[];
    _Float16* Mh = (_Float16*)smem;                              // [96][256]
    _Float16* Mt = (_Float16*)(smem + 96*256*2);                 // [256][96]
    _Float16* Pb = (_Float16*)(smem + 96*256*2 + 256*96*2);      // [5][16][96]

    const int bt = blockIdx.x;
    const int b = bt / T_, t = bt % T_;

    // stage M (73 rows + zero pad to 96) in both layouts
    for (int i = threadIdx.x; i < 96 * 256; i += 160) {
        int row = i >> 8, d = i & 255;
        float v = 0.f;
        if (row < 72) {
            int s = row / C_, c = row % C_;
            v = satt[((size_t)(s * B_ + b) * N_ + ((size_t)t * C_ + c)) * D_ + d];
        } else if (row == 72) {
            v = tf32[(size_t)bt * D_ + d];
        }
        _Float16 hv = (_Float16)v;
        Mh[row * 256 + d] = hv;
        Mt[d * 96 + row]  = hv;
    }
    for (int i = threadIdx.x; i < 5 * 16 * 96; i += 160) Pb[i] = (_Float16)0.f;
    // zero the never-written transposed-output pad columns 80..95
    for (int i = threadIdx.x; i < 256 * 16; i += 160) {
        int d = i >> 4, k = 80 + (i & 15);
        mout_t[((size_t)bt * 256 + d) * 96 + k] = (_Float16)0.f;
    }
    __syncthreads();

    const int lane = threadIdx.x & 31;
    const int wave = threadIdx.x >> 5;        // 0..4
    const int lm = lane & 15, lg = lane >> 4;
    const int q0 = wave * 16;
    _Float16* Pw = Pb + wave * 16 * 96;

    // Q fragments (A layout) from LDS
    v16h qf[8];
    {
        const _Float16* qrow = Mh + (q0 + lm) * 256;
#pragma unroll
        for (int c = 0; c < 8; c++) {
            v8h lo = *(const v8h*)(qrow + 32 * c + lg * 16);
            v8h hi = *(const v8h*)(qrow + 32 * c + lg * 16 + 8);
            qf[c] = __builtin_shufflevector(lo, hi, 0,1,2,3,4,5,6,7,8,9,10,11,12,13,14,15);
        }
    }
    // scores: 5 key tiles of 16
    v8f s[5];
#pragma unroll
    for (int kt = 0; kt < 5; kt++) {
        v8f z = {0.f,0.f,0.f,0.f,0.f,0.f,0.f,0.f};
        s[kt] = z;
#pragma unroll
        for (int c = 0; c < 8; c++) {
            v16h bf = *(const v16h*)(Mh + (kt * 16 + lm) * 256 + 32 * c + lg * 16);
            s[kt] = __builtin_amdgcn_wmma_f32_16x16x32_f16(false, qf[c], false, bf, (short)0, s[kt], false, false);
        }
    }
    // softmax (scale, mask keys >= 73)
    float invl[8];
    float pv[5][8];
#pragma unroll
    for (int r = 0; r < 8; r++) {
        float mx = -1e30f;
#pragma unroll
        for (int kt = 0; kt < 5; kt++) {
            float v = s[kt][r] * SCALE_;
            if (kt == 4 && lm >= 9) v = -1e30f;
            pv[kt][r] = v;
            mx = fmaxf(mx, v);
        }
#pragma unroll
        for (int off = 1; off < 16; off <<= 1) mx = fmaxf(mx, __shfl_xor(mx, off, 32));
        float den = 0.f;
#pragma unroll
        for (int kt = 0; kt < 5; kt++) { float e = __expf(pv[kt][r] - mx); pv[kt][r] = e; den += e; }
#pragma unroll
        for (int off = 1; off < 16; off <<= 1) den += __shfl_xor(den, off, 32);
        invl[r] = 1.f / den;
    }
    // bounce P (16x96, cols 80..95 pre-zeroed) to A layout
#pragma unroll
    for (int kt = 0; kt < 5; kt++)
#pragma unroll
        for (int r = 0; r < 8; r++)
            Pw[(r + 8 * lg) * 96 + kt * 16 + lm] = (_Float16)pv[kt][r];
    __syncthreads();
    v16h pf[3];
#pragma unroll
    for (int c = 0; c < 3; c++) {
        v8h lo = *(const v8h*)(Pw + lm * 96 + 32 * c + lg * 16);
        v8h hi = *(const v8h*)(Pw + lm * 96 + 32 * c + lg * 16 + 8);
        pf[c] = __builtin_shufflevector(lo, hi, 0,1,2,3,4,5,6,7,8,9,10,11,12,13,14,15);
    }
    // O = P * M ; write row-major and transposed
#pragma unroll
    for (int j = 0; j < 16; j++) {
        v8f acc = {0.f,0.f,0.f,0.f,0.f,0.f,0.f,0.f};
#pragma unroll
        for (int c = 0; c < 3; c++) {
            v16h vb = *(const v16h*)(Mt + (16 * j + lm) * 96 + 32 * c + lg * 16);
            acc = __builtin_amdgcn_wmma_f32_16x16x32_f16(false, pf[c], false, vb, (short)0, acc, false, false);
        }
        int d = 16 * j + lm;
        v8h pk;
#pragma unroll
        for (int r = 0; r < 8; r++) {
            int row = q0 + 8 * lg + r;
            float val = (row < CT_) ? acc[r] * invl[r] : 0.f;
            _Float16 hv = (_Float16)val;
            pk[r] = hv;
            mout_h[((size_t)bt * 80 + row) * 256 + d] = hv;
        }
        *(v8h*)(mout_t + ((size_t)bt * 256 + d) * 96 + q0 + 8 * lg) = pk;
    }
}

// ---------------------------------------------------------------------------
// K4b: med-query attention (no scale) via WMMA; 2 waves (Q rows 0..31,
//      rows >=24 discarded); sum over the 24 query rows; + bias + pos emb.
// ---------------------------------------------------------------------------
__global__ __launch_bounds__(64)
void k4b_med_out(const _Float16* __restrict__ mout_h, const _Float16* __restrict__ mout_t,
                 const int* __restrict__ input_len, const float* __restrict__ bias_med,
                 float* __restrict__ outp)
{
    __shared__ __align__(64) _Float16 Pb[2 * 16 * 96];
    __shared__ float wsum[2 * 256];

    const int bt = blockIdx.x;
    const int b = bt / T_, t = bt % T_;
    for (int i = threadIdx.x; i < 2 * 16 * 96; i += 64) Pb[i] = (_Float16)0.f;
    __syncthreads();

    const int lane = threadIdx.x & 31, wave = threadIdx.x >> 5; // 0..1
    const int lm = lane & 15, lg = lane >> 4;
    const int q0 = wave * 16;
    _Float16* Pw = Pb + wave * 16 * 96;
    const _Float16* Mb  = mout_h + (size_t)bt * 80 * 256;
    const _Float16* Mtb = mout_t + (size_t)bt * 256 * 96;

    v16h qf[8];
    {
        const _Float16* qrow = Mb + (q0 + lm) * 256;
#pragma unroll
        for (int c = 0; c < 8; c++) {
            v8h lo = *(const v8h*)(qrow + 32 * c + lg * 16);
            v8h hi = *(const v8h*)(qrow + 32 * c + lg * 16 + 8);
            qf[c] = __builtin_shufflevector(lo, hi, 0,1,2,3,4,5,6,7,8,9,10,11,12,13,14,15);
        }
    }
    v8f s[5];
#pragma unroll
    for (int kt = 0; kt < 5; kt++) {
        v8f z = {0.f,0.f,0.f,0.f,0.f,0.f,0.f,0.f};
        s[kt] = z;
#pragma unroll
        for (int c = 0; c < 8; c++) {
            v16h bf = *(const v16h*)(Mb + (kt * 16 + lm) * 256 + 32 * c + lg * 16);
            s[kt] = __builtin_amdgcn_wmma_f32_16x16x32_f16(false, qf[c], false, bf, (short)0, s[kt], false, false);
        }
    }
    float invl[8];
    float pv[5][8];
#pragma unroll
    for (int r = 0; r < 8; r++) {
        float mx = -1e30f;
#pragma unroll
        for (int kt = 0; kt < 5; kt++) {
            float v = s[kt][r];                   // reference: no scale here
            if (kt == 4 && lm >= 9) v = -1e30f;   // mask keys >= 73
            pv[kt][r] = v;
            mx = fmaxf(mx, v);
        }
#pragma unroll
        for (int off = 1; off < 16; off <<= 1) mx = fmaxf(mx, __shfl_xor(mx, off, 32));
        float den = 0.f;
#pragma unroll
        for (int kt = 0; kt < 5; kt++) { float e = __expf(pv[kt][r] - mx); pv[kt][r] = e; den += e; }
#pragma unroll
        for (int off = 1; off < 16; off <<= 1) den += __shfl_xor(den, off, 32);
        invl[r] = 1.f / den;
    }
#pragma unroll
    for (int kt = 0; kt < 5; kt++)
#pragma unroll
        for (int r = 0; r < 8; r++)
            Pw[(r + 8 * lg) * 96 + kt * 16 + lm] = (_Float16)pv[kt][r];
    __syncthreads();
    v16h pf[3];
#pragma unroll
    for (int c = 0; c < 3; c++) {
        v8h lo = *(const v8h*)(Pw + lm * 96 + 32 * c + lg * 16);
        v8h hi = *(const v8h*)(Pw + lm * 96 + 32 * c + lg * 16 + 8);
        pf[c] = __builtin_shufflevector(lo, hi, 0,1,2,3,4,5,6,7,8,9,10,11,12,13,14,15);
    }
    // O3 = P * O2 ; fold the sum over the 24 valid query rows
#pragma unroll
    for (int j = 0; j < 16; j++) {
        v8f acc = {0.f,0.f,0.f,0.f,0.f,0.f,0.f,0.f};
#pragma unroll
        for (int c = 0; c < 3; c++) {
            v16h vb = *(const v16h*)(Mtb + (16 * j + lm) * 96 + 32 * c + lg * 16);
            acc = __builtin_amdgcn_wmma_f32_16x16x32_f16(false, pf[c], false, vb, (short)0, acc, false, false);
        }
        float partial = 0.f;
        if (wave == 0 || lg == 0) {               // rows 0..15 / 16..23 valid
#pragma unroll
            for (int r = 0; r < 8; r++) partial += acc[r] * invl[r];
        }
        partial += __shfl_xor(partial, 16, 32);
        wsum[wave * 256 + 16 * j + lm] = partial;
    }
    __syncthreads();

#pragma unroll
    for (int k = 0; k < 4; k++) {
        int d = threadIdx.x + 64 * k;
        float tot = wsum[d] + wsum[256 + d];
        float pe = 0.f;
        if (t + 1 <= input_len[b]) {
            double expo  = (double)(2 * (d / 2)) / 256.0;
            double angle = (double)t / pow(10000.0, expo);
            pe = (float)((d & 1) ? cos(angle) : sin(angle));
        }
        outp[(size_t)bt * D_ + d] = tot + bias_med[d] + pe;
    }
}

// ---------------------------------------------------------------------------
extern "C" void kernel_launch(void* const* d_in, const int* in_sizes, int n_in,
                              void* d_out, int out_size, void* d_ws, size_t ws_size,
                              hipStream_t stream)
{
    const int*   med_codes  = (const int*)  d_in[0];
    const int*   lab_codes  = (const int*)  d_in[1];
    const int*   diag_codes = (const int*)  d_in[2];
    /* d_in[3] = mask, unused by forward */
    const float* med_mask   = (const float*)d_in[4];
    const float* lab_mask   = (const float*)d_in[5];
    const float* diag_mask  = (const float*)d_in[6];
    const float* seq_time   = (const float*)d_in[7];
    const int*   input_len  = (const int*)  d_in[8];
    const float* emb_med    = (const float*)d_in[9];
    const float* emb_lab    = (const float*)d_in[10];
    const float* emb_diag   = (const float*)d_in[11];
    const float* bias_med   = (const float*)d_in[12];
    const float* W_sel      = (const float*)d_in[15];
    const float* b_sel      = (const float*)d_in[16];
    const float* W_time     = (const float*)d_in[17];
    const float* b_time     = (const float*)d_in[18];
    const float* Wg_med     = (const float*)d_in[19];
    const float* bg_med     = (const float*)d_in[20];
    const float* Wg_lab     = (const float*)d_in[21];
    const float* bg_lab     = (const float*)d_in[22];
    const float* Wg_diag    = (const float*)d_in[23];
    const float* bg_diag    = (const float*)d_in[24];
    float* outp = (float*)d_out;

    // workspace layout
    const size_t TF_BYTES   = (size_t)B_ * T_ * D_ * sizeof(float);        //  1.3 MB
    const size_t GH_BYTES   = (size_t)3 * B_ * N_ * D_ * sizeof(_Float16); // 47.2 MB
    const size_t SATT_BYTES = (size_t)3 * B_ * N_ * D_ * sizeof(float);    // 94.4 MB
    char* ws = (char*)d_ws;
    float*    tf32    = (float*)ws;
    _Float16* gated_h = (_Float16*)(ws + TF_BYTES);
    _Float16* gated_t = (_Float16*)(ws + TF_BYTES + GH_BYTES);
    float*    satt    = (float*)   (ws + TF_BYTES + 2 * GH_BYTES);
    // mout_h (52.4 MB, 80-row pitch) reuses the gated region (dead after K3)
    _Float16* mout_h  = (_Float16*)(ws + TF_BYTES);
    // mout_t (62.9 MB, 96-key pitch) after satt
    _Float16* mout_t  = (_Float16*)(ws + TF_BYTES + 2 * GH_BYTES + SATT_BYTES);
    (void)in_sizes; (void)n_in; (void)out_size; (void)ws_size;

    k1_timefeat<<<dim3(B_ * T_), dim3(64), 0, stream>>>(seq_time, W_sel, b_sel, W_time, b_time, tf32);

    k2_gate<<<dim3(B_ * T_, 3), dim3(256), 0, stream>>>(
        med_codes, lab_codes, diag_codes, med_mask, lab_mask, diag_mask,
        emb_med, emb_lab, emb_diag, Wg_med, bg_med, Wg_lab, bg_lab, Wg_diag, bg_diag,
        tf32, gated_h, gated_t);

    k3_stream_attn<<<dim3(N_ / 16, 3 * B_), dim3(32), 0, stream>>>(gated_h, gated_t, satt);

    k4a_merge_attn<<<dim3(B_ * T_), dim3(160), K4A_LDS, stream>>>(satt, tf32, mout_h, mout_t);

    k4b_med_out<<<dim3(B_ * T_), dim3(64), 0, stream>>>(mout_h, mout_t, input_len, bias_med, outp);
}
